// GoSpecificWattiPooling_20332375179843
// MI455X (gfx1250) — matrix operand order repaired and checked by reading, not compiled, not run on hardware
//
#include <hip/hip_runtime.h>

// ---------------- problem constants ----------------
#define B_   16
#define L_   2048
#define T_   512
#define DH   1280
#define DG   768
#define DP   256
#define SCALE_ 0.0625f     // 256^-0.5
#define LC   32            // L chunk per iteration
#define NCH  (L_ / LC)     // 64 chunks
#define DW   160           // Dh columns per wave (8 waves * 160 = 1280)

// LDS map (dynamic, attn kernel)
#define QS_OFF   0u        // 16 x 256 bf16 (8 KB)
#define KS0_OFF  8192u     // 32 x 256 bf16 (16 KB), buffer 0
#define KS1_OFF  24576u    // 32 x 256 bf16 (16 KB), buffer 1
#define PS_OFF   40960u    // 16 x 34 bf16
#define SCL_OFF  42048u    // float[16]
#define SINV_OFF 42112u    // float[16]
#define HS_OFF   42176u    // 1280 x 34 bf16 (transposed H chunk, 87040 B)
#define SMEM_SZ  (42176u + 1280u * 34u * 2u)   // 129,216 bytes

// ---------------- CDNA5 WMMA types ----------------
typedef __attribute__((ext_vector_type(16))) __bf16 v16bf;
typedef __attribute__((ext_vector_type(8)))  float  v8f;
typedef __attribute__((ext_vector_type(4)))  unsigned v4u;
typedef __attribute__((ext_vector_type(8)))  unsigned v8u;

union Frag { unsigned u[8]; v16bf v; };

__device__ __forceinline__ v8f wmma_bf16(const Frag a, const Frag b, v8f c) {
  // D = A(16x32 bf16) * B(32x16 bf16) + C(16x16 f32)
  return __builtin_amdgcn_wmma_f32_16x16x32_bf16(false, a.v, false, b.v,
                                                 (short)0, c, false, false);
}

__device__ __forceinline__ unsigned short f2bf(float f) {
  unsigned u = __float_as_uint(f);
  u += 0x7FFFu + ((u >> 16) & 1u);   // round-to-nearest-even
  return (unsigned short)(u >> 16);
}

__device__ __forceinline__ v8f zero8() {
  v8f z;
#pragma unroll
  for (int i = 0; i < 8; ++i) z[i] = 0.0f;
  return z;
}

// ---- gfx1250 async global->LDS copy, 16 bytes per lane (GVS form) ----
__device__ __forceinline__ void async_copy16(unsigned lds_off, const void* sbase,
                                             unsigned voff, int ioff) {
  asm volatile("global_load_async_to_lds_b128 %0, %1, %2 offset:%3"
               :: "v"(lds_off), "v"(voff), "s"(sbase), "i"(ioff)
               : "memory");
}

// ---- gfx1250 Tensor Data Mover: 32 x 256 bf16 tile, contiguous rows ----
__device__ __forceinline__ void tdm_load_k256(unsigned lds_byte_off,
                                              const void* gptr) {
  const unsigned long long ga = (unsigned long long)(size_t)gptr;
  v4u g0;
  g0[0] = 1u;                                   // count=1, user descriptor
  g0[1] = lds_byte_off;                         // lds_addr (bytes)
  g0[2] = (unsigned)ga;                         // global_addr[31:0]
  g0[3] = ((unsigned)(ga >> 32) & 0x01FFFFFFu)  // global_addr[56:32]
          | 0x80000000u;                        // type=2 ("image")
  v8u g1;
  g1[0] = 0x00010000u;   // workgroup_mask=0, data_size=1 (2 bytes)
  g1[1] = 0x01000000u;   // tensor_dim0 = 256  (bits 79:48, low 16 in [31:16])
  g1[2] = 0x00200000u;   // tensor_dim0 hi = 0; tensor_dim1 = 32 (bits 111:80)
  g1[3] = 0x01000000u;   // tensor_dim1 hi = 0; tile_dim0 = 256 (bits 127:112)
  g1[4] = 32u;           // tile_dim1 = 32; tile_dim2 = 0
  g1[5] = 256u;          // tensor_dim0_stride = 256 (bits 207:160)
  g1[6] = 0u;
  g1[7] = 0u;
  asm volatile("tensor_load_to_lds %0, %1" :: "s"(g0), "s"(g1) : "memory");
}

// =====================================================================
// Projection: Out[r, p] = sum_d X[r, d] * W[p, d], bf16 output.
// block = 128 threads (4 waves); grid.x = rows/16, grid.y = 256/64.
// =====================================================================
__global__ __launch_bounds__(128) void proj_bf16_kernel(
    const float* __restrict__ X, const float* __restrict__ W,
    unsigned short* __restrict__ Out, int dIn)
{
  __shared__ unsigned short xs[16 * 34];
  __shared__ unsigned short wsm[64 * 34];

  const int tid  = threadIdx.x;
  const int wv   = tid >> 5;
  const int lane = tid & 31;
  const int li   = lane & 15;
  const int hh   = lane >> 4;
  const long r0  = (long)blockIdx.x * 16;
  const int  cb  = blockIdx.y * 64;

  v8f acc = zero8();
  const int nk = dIn >> 5;

  for (int kc = 0; kc < nk; ++kc) {
    const int k0 = kc * 32;
    {
      const int l = tid >> 3, d = (tid & 7) * 4;
      const float4 v = *(const float4*)(X + (r0 + l) * dIn + k0 + d);
      xs[l * 34 + d + 0] = f2bf(v.x);
      xs[l * 34 + d + 1] = f2bf(v.y);
      xs[l * 34 + d + 2] = f2bf(v.z);
      xs[l * 34 + d + 3] = f2bf(v.w);
    }
#pragma unroll
    for (int jj = 0; jj < 4; ++jj) {
      const int ff = jj * 128 + tid;
      const int r = ff >> 3, d = (ff & 7) * 4;
      const float4 v = *(const float4*)(W + (long)(cb + r) * dIn + k0 + d);
      wsm[r * 34 + d + 0] = f2bf(v.x);
      wsm[r * 34 + d + 1] = f2bf(v.y);
      wsm[r * 34 + d + 2] = f2bf(v.z);
      wsm[r * 34 + d + 3] = f2bf(v.w);
    }
    __syncthreads();

    Frag a, b;
#pragma unroll
    for (int j = 0; j < 8; ++j) {
      const int kj = 16 * (j >> 2) + 2 * (j & 3) + 8 * hh;   // A layout
      a.u[j] = *(const unsigned*)&xs[li * 34 + kj];
      b.u[j] = *(const unsigned*)&wsm[(wv * 16 + li) * 34 + kj];
    }
    acc = wmma_bf16(a, b, acc);
    __syncthreads();
  }

#pragma unroll
  for (int g = 0; g < 8; ++g) {
    const long row = r0 + g + 8 * hh;
    const int  col = cb + wv * 16 + li;
    Out[row * DP + col] = f2bf(acc[g]);
  }
}

// =====================================================================
// Fused attention: one workgroup = (batch b, 16-row T tile), 8 wave32s.
// K chunks double-buffered via TDM (tensor_load_to_lds), chunk n+1 in
// flight while chunk n is consumed; s_wait_tensorcnt<=1 retires oldest.
// =====================================================================
__global__ __launch_bounds__(256) void attn_kernel(
    const float* __restrict__ H,
    const unsigned short* __restrict__ Qb,
    const unsigned short* __restrict__ Kb,
    float* __restrict__ Out)
{
  extern __shared__ unsigned char smem[];
  unsigned short* qs     = (unsigned short*)(smem + QS_OFF);
  unsigned short* ps     = (unsigned short*)(smem + PS_OFF);
  float*          scaleS = (float*)(smem + SCL_OFF);
  float*          sinvS  = (float*)(smem + SINV_OFF);
  unsigned short* hs     = (unsigned short*)(smem + HS_OFF);

  const int tid  = threadIdx.x;
  const int wv   = tid >> 5;
  const int lane = tid & 31;
  const int li   = lane & 15;
  const int hh   = lane >> 4;
  const int tt   = blockIdx.x;   // T tile [0,32)
  const int b    = blockIdx.y;   // batch

  const unsigned short* kbase = Kb + (long)b * L_ * DP;
  const float*          hbase = H  + (long)b * L_ * DH;

  // ---- async copy Q tile (16x256 bf16 = 8 KB) into LDS ----
  {
    const void* qsrc = (const void*)(Qb + (long)(b * T_ + tt * 16) * DP);
    const unsigned off16 = (unsigned)tid * 16u;
    async_copy16(off16, qsrc, off16, 0);
    async_copy16(off16, qsrc, off16, 4096);
  }
  // ---- TDM: prologue load of K chunk 0 into buffer 0 ----
  if (wv == 0) tdm_load_k256(KS0_OFF, kbase);
  asm volatile("s_wait_asynccnt 0x0" ::: "memory");

  float m = -3.0e38f, sden = 0.0f;   // online-softmax state (wave0; t = li)
  v8f acc[10];
#pragma unroll
  for (int nt = 0; nt < 10; ++nt) acc[nt] = zero8();
  const int wd0 = wv * DW;

  for (int lcb = 0; lcb < NCH; ++lcb) {
    const int l0 = lcb * LC;
    const unsigned short* ksc =
        (const unsigned short*)(smem + ((lcb & 1) ? KS1_OFF : KS0_OFF));

    // ---- TDM: launch K chunk n+1 into the other buffer ----
    if (wv == 0 && lcb + 1 < NCH)
      tdm_load_k256((lcb & 1) ? KS0_OFF : KS1_OFF,
                    kbase + (long)(l0 + LC) * DP);

    if (lcb + 1 < NCH)   // prefetch next H chunk -> global_prefetch_b8
      __builtin_prefetch(hbase + (long)(l0 + LC) * DH + (tid << 6), 0, 0);

    // ---- stage H chunk 32 x 1280 f32 -> transposed bf16 hs[d][l] ----
    for (int j = 0; j < 40; ++j) {
      const int ff = j * 256 + tid;          // float4 index over [32][320]
      const int l  = ff / 320;
      const int d  = (ff % 320) * 4;
      const float4 v = *(const float4*)(hbase + (long)(l0 + l) * DH + d);
      hs[(d + 0) * 34 + l] = f2bf(v.x);
      hs[(d + 1) * 34 + l] = f2bf(v.y);
      hs[(d + 2) * 34 + l] = f2bf(v.z);
      hs[(d + 3) * 34 + l] = f2bf(v.w);
    }
    // retire the TDM for THIS chunk; next chunk may stay in flight
    if (wv == 0) {
      if (lcb + 1 < NCH) __builtin_amdgcn_s_wait_tensorcnt(1);
      else               __builtin_amdgcn_s_wait_tensorcnt(0);
    }
    __syncthreads();

    // ---- wave0: S^T = K*Q^T (32l x 16t), online softmax ----
    if (wv == 0) {
      v8f c0 = zero8(), c1 = zero8();
#pragma unroll
      for (int ksr = 0; ksr < 8; ++ksr) {        // P=256 -> 8 k-steps of 32
        Frag a0, a1, qb;
#pragma unroll
        for (int j = 0; j < 8; ++j) {
          const int kj = ksr * 32 + 16 * (j >> 2) + 2 * (j & 3) + 8 * hh;
          a0.u[j] = *(const unsigned*)&ksc[li * DP + kj];          // l 0..15
          a1.u[j] = *(const unsigned*)&ksc[(16 + li) * DP + kj];   // l 16..31
          qb.u[j] = *(const unsigned*)&qs[li * DP + kj];
        }
        c0 = wmma_bf16(a0, qb, c0);
        c1 = wmma_bf16(a1, qb, c1);
      }
      // lane holds logits[t=li][ l = g+8*hh (c0), 16+g+8*hh (c1) ]
      float v0[8], v1[8], cmax = -3.0e38f;
#pragma unroll
      for (int g = 0; g < 8; ++g) {
        v0[g] = c0[g] * SCALE_;
        v1[g] = c1[g] * SCALE_;
        cmax = fmaxf(cmax, fmaxf(v0[g], v1[g]));
      }
      cmax = fmaxf(cmax, __shfl_xor(cmax, 16, 32));
      const float mnew = fmaxf(m, cmax);
      const float al   = __expf(m - mnew);
      float sum = 0.0f;
#pragma unroll
      for (int g = 0; g < 8; ++g) {
        v0[g] = __expf(v0[g] - mnew);
        v1[g] = __expf(v1[g] - mnew);
        sum += v0[g] + v1[g];
      }
      sum += __shfl_xor(sum, 16, 32);
      sden = sden * al + sum;
      m = mnew;
#pragma unroll
      for (int g = 0; g < 8; ++g) {
        ps[li * 34 + (g + 8 * hh)]      = f2bf(v0[g]);
        ps[li * 34 + (16 + g + 8 * hh)] = f2bf(v1[g]);
      }
      if (hh == 0) scaleS[li] = al;
    }
    __syncthreads();

    // ---- all waves: rescale accumulators, Z += P * H ----
    float al8[8];
#pragma unroll
    for (int g = 0; g < 8; ++g) al8[g] = scaleS[g + 8 * hh];
    Frag pa;
#pragma unroll
    for (int j = 0; j < 8; ++j) {
      const int kj = 16 * (j >> 2) + 2 * (j & 3) + 8 * hh;
      pa.u[j] = *(const unsigned*)&ps[li * 34 + kj];
    }
#pragma unroll
    for (int nt = 0; nt < 10; ++nt) {
      Frag hb;
      const int d = wd0 + nt * 16 + li;
#pragma unroll
      for (int j = 0; j < 8; ++j)
        hb.u[j] = *(const unsigned*)&hs[d * 34 + 16 * (j >> 2) + 2 * (j & 3) + 8 * hh];
      v8f t = acc[nt];
#pragma unroll
      for (int g = 0; g < 8; ++g) t[g] *= al8[g];
      acc[nt] = wmma_bf16(pa, hb, t);
    }
    __syncthreads();
  }

  // ---- epilogue: divide by softmax denominator, write f32 output ----
  if (wv == 0 && hh == 0) sinvS[li] = 1.0f / sden;
  __syncthreads();
  float si[8];
#pragma unroll
  for (int g = 0; g < 8; ++g) si[g] = sinvS[g + 8 * hh];
  float* obase = Out + (long)(b * T_ + tt * 16) * DH;
#pragma unroll
  for (int nt = 0; nt < 10; ++nt) {
    const int d = wd0 + nt * 16 + li;
#pragma unroll
    for (int g = 0; g < 8; ++g)
      obase[(long)(g + 8 * hh) * DH + d] = acc[nt][g] * si[g];
  }
}

// =====================================================================
extern "C" void kernel_launch(void* const* d_in, const int* in_sizes, int n_in,
                              void* d_out, int out_size, void* d_ws, size_t ws_size,
                              hipStream_t stream) {
  (void)in_sizes; (void)n_in; (void)out_size; (void)ws_size;
  const float* H  = (const float*)d_in[0];   // [16,2048,1280]
  const float* G  = (const float*)d_in[1];   // [16,512,768]
  const float* Wq = (const float*)d_in[2];   // [256,768]
  const float* Wk = (const float*)d_in[3];   // [256,1280]
  float* Z = (float*)d_out;                  // [16,512,1280]

  unsigned short* Qbuf = (unsigned short*)d_ws;            // 8192 x 256 bf16
  unsigned short* Kbuf = Qbuf + (size_t)(B_ * T_) * DP;    // 32768 x 256 bf16

  proj_bf16_kernel<<<dim3((B_ * T_) / 16, DP / 64), 128, 0, stream>>>(G, Wq, Qbuf, DG);
  proj_bf16_kernel<<<dim3((B_ * L_) / 16, DP / 64), 128, 0, stream>>>(H, Wk, Kbuf, DH);

  attn_kernel<<<dim3(T_ / 16, B_), 256, SMEM_SZ, stream>>>(H, Qbuf, Kbuf, Z);
}
